// SpikeFP16Linear_MultiPrecision_43860206027351
// MI455X (gfx1250) — compile-verified, matrix operand order checked by reading, not run on hardware
//
#include <hip/hip_runtime.h>

typedef __attribute__((ext_vector_type(16))) _Float16 v16h;
typedef __attribute__((ext_vector_type(8)))  _Float16 v8h;
typedef __attribute__((ext_vector_type(8)))  float    v8f;
typedef __attribute__((ext_vector_type(4)))  float    f4;

// ---------------------------------------------------------------------------
// Pass 1: decode 16 float32 bit-planes -> one fp16 word. Pure bandwidth
// (reads 64B, writes 2B per element). Vectorized as 4x b128 loads per thread.
// ---------------------------------------------------------------------------
__global__ __launch_bounds__(256)
void decode_pulse_fp16(const float* __restrict__ bits,
                       _Float16* __restrict__ out, int n) {
    int i = blockIdx.x * blockDim.x + threadIdx.x;
    if (i >= n) return;
    const f4* p = (const f4*)(bits + (size_t)i * 16);
    unsigned u = 0;
#pragma unroll
    for (int j = 0; j < 4; ++j) {
        f4 b = p[j];
        u |= ((unsigned)b.x) << (4 * j + 0);
        u |= ((unsigned)b.y) << (4 * j + 1);
        u |= ((unsigned)b.z) << (4 * j + 2);
        u |= ((unsigned)b.w) << (4 * j + 3);
    }
    unsigned short us = (unsigned short)u;
    out[i] = __builtin_bit_cast(_Float16, us);
}

// ---------------------------------------------------------------------------
// Pass 2: WMMA GEMM  acc[m][n] = sum_k X[m][k] * W[n][k]   (f16 x f16 -> f32)
// One 16x16 tile per wave32; K=512 -> 16x v_wmma_f32_16x16x32_f16.
// Epilogue fused: f32 -> fp16 (RNE) -> 16 bit-planes, 4x b128 stores/element.
// ---------------------------------------------------------------------------
__global__ __launch_bounds__(256)
void gemm_wmma_encode(const _Float16* __restrict__ X,   // [M][K]
                      const _Float16* __restrict__ W,   // [N][K]
                      float* __restrict__ outBits,      // [M][N][16]
                      int M, int N, int K) {
    const int lane = threadIdx.x & 31;
    const int wave = threadIdx.x >> 5;            // wave32
    const int nTiles = N >> 4;
    const int tile = blockIdx.x * 8 + wave;
    const int mT = tile / nTiles;
    const int nT = tile % nTiles;
    if (mT * 16 >= M) return;                     // wave-uniform exit

    const int r  = lane & 15;                     // A row / B column within tile
    const int kh = (lane >> 4) << 3;              // K phase: 0 (lanes 0-15) or 8

    const _Float16* xp = X + (size_t)(mT * 16 + r) * K;
    const _Float16* wp = W + (size_t)(nT * 16 + r) * K;

    v8f acc = {};
#pragma unroll 4
    for (int kb = 0; kb < K; kb += 32) {
        // A 16x32 fragment: lane holds K = kb+kh..+7 and kb+16+kh..+7
        v8h alo = *(const v8h*)(xp + kb + kh);
        v8h ahi = *(const v8h*)(xp + kb + 16 + kh);
        // B 32x16 fragment: same K striping over w's row (w is [N][K])
        v8h blo = *(const v8h*)(wp + kb + kh);
        v8h bhi = *(const v8h*)(wp + kb + 16 + kh);
        v16h a = __builtin_shufflevector(alo, ahi,
                 0,1,2,3,4,5,6,7,8,9,10,11,12,13,14,15);
        v16h b = __builtin_shufflevector(blo, bhi,
                 0,1,2,3,4,5,6,7,8,9,10,11,12,13,14,15);
        acc = __builtin_amdgcn_wmma_f32_16x16x32_f16(
            /*neg_a=*/false, a, /*neg_b=*/false, b,
            /*c_mod=*/(short)0, acc, /*reuse_a=*/false, /*reuse_b=*/false);
    }

    // C/D layout: VGPR i -> M = i (lanes 0-15) or i+8 (lanes 16-31); N = lane&15
    const int mBase = mT * 16 + ((lane >> 4) << 3);
    const int col   = nT * 16 + r;
#pragma unroll
    for (int i = 0; i < 8; ++i) {
        _Float16 h = (_Float16)acc[i];            // v_cvt_f16_f32, RNE
        unsigned u = (unsigned)__builtin_bit_cast(unsigned short, h);
        float* op = outBits + (((size_t)(mBase + i) * N) + col) * 16;
#pragma unroll
        for (int g = 0; g < 4; ++g) {
            f4 o;
            o.x = (float)((u >> (4 * g + 0)) & 1u);
            o.y = (float)((u >> (4 * g + 1)) & 1u);
            o.z = (float)((u >> (4 * g + 2)) & 1u);
            o.w = (float)((u >> (4 * g + 3)) & 1u);
            *(f4*)(op + 4 * g) = o;
        }
    }
}

extern "C" void kernel_launch(void* const* d_in, const int* in_sizes, int n_in,
                              void* d_out, int out_size, void* d_ws, size_t ws_size,
                              hipStream_t stream) {
    const float* x_pulse = (const float*)d_in[0];   // [B,T,IN,16] f32
    const float* w_pulse = (const float*)d_in[1];   // [OUT,IN,16] f32

    const int IN = 512, OUT = 512;
    const int xElems = in_sizes[0] / 16;            // B*T*IN  = 524288
    const int wElems = in_sizes[1] / 16;            // OUT*IN  = 262144
    const int M = xElems / IN;                      // B*T     = 1024

    _Float16* xdec = (_Float16*)d_ws;               // 1.0 MB
    _Float16* wdec = xdec + xElems;                 // 0.5 MB  (fits any ws)

    decode_pulse_fp16<<<(xElems + 255) / 256, 256, 0, stream>>>(x_pulse, xdec, xElems);
    decode_pulse_fp16<<<(wElems + 255) / 256, 256, 0, stream>>>(w_pulse, wdec, wElems);

    const int totalTiles = (M / 16) * (OUT / 16);   // 2048 tiles, 8 waves/block
    gemm_wmma_encode<<<totalTiles / 8, 256, 0, stream>>>(xdec, wdec,
                                                         (float*)d_out, M, OUT, IN);
}
